// LSTM_36807869726767
// MI455X (gfx1250) — matrix operand order repaired and checked by reading, not compiled, not run on hardware
//
#include <hip/hip_runtime.h>
#include <hip/hip_bf16.h>

// ---------------------------------------------------------------------------
// LSTM + attention classifier for MI455X (gfx1250, wave32, WMMA).
// B=64, S=512, E=512, H=1024, 4H=4096, K_in=E+H=1536, NCLS=10.
// ---------------------------------------------------------------------------

typedef __attribute__((ext_vector_type(16))) __bf16 v16bf;
typedef __attribute__((ext_vector_type(8)))  float  v8f;
typedef __attribute__((ext_vector_type(4)))  unsigned int u32x4;

union ABFrag { v16bf v; u32x4 q[2]; };

#define WMMA_BF16(A_, B_, C_) \
  __builtin_amdgcn_wmma_f32_16x16x32_bf16(false, (A_), false, (B_), (short)0, (C_), false, false)

#define NB 64
#define NS 512
#define NE 512
#define NH 1024
#define NG 4096
#define NKIN 1536

// ------------------------------- helpers -----------------------------------

__global__ void k_f32_to_bf16(const float* __restrict__ src,
                              __bf16* __restrict__ dst, int n) {
  int i = blockIdx.x * blockDim.x + threadIdx.x;
  if (i < n) dst[i] = (__bf16)src[i];
}

// gather embeddings -> bf16  [B,S,E]
__global__ void k_embed_gather(const int* __restrict__ x,
                               const float* __restrict__ emb,
                               __bf16* __restrict__ ebf, int n) {
  int i = blockIdx.x * blockDim.x + threadIdx.x;
  if (i >= n) return;
  int k  = i & (NE - 1);
  int bs = i >> 9;                 // NE == 512
  int tok = x[bs];
  ebf[i] = (__bf16)emb[(size_t)tok * NE + k];
}

__global__ void k_init_state(float* __restrict__ c,
                             __bf16* __restrict__ hbf, int n) {
  int i = blockIdx.x * blockDim.x + threadIdx.x;
  if (i < n) { c[i] = 0.0f; hbf[i] = (__bf16)0.0f; }
}

// --------------------------- recurrent GEMM --------------------------------
// gates[B,4H] = concat(e_t, h) @ Wg^T      (bias added in cell kernel)
// One wave -> 16x64 output tile; 4 accumulators share one A fragment.
__global__ __launch_bounds__(256) void k_lstm_gates(
    const __bf16* __restrict__ ebf,   // [B,S,E]
    const __bf16* __restrict__ hbf,   // [B,H]
    const __bf16* __restrict__ wg,    // [4H, 1536] row-major (bf16)
    float* __restrict__ gates,        // [B,4H]
    int t) {
  const int lane = threadIdx.x & 31;
  const int half = lane >> 4;
  const int l16  = lane & 15;
  const int wave = blockIdx.x * (blockDim.x >> 5) + (threadIdx.x >> 5); // 0..255
  const int mtile = wave & 3;        // B/16 = 4
  const int nbase = (wave >> 2) * 64; // 4096/64 = 64 quads

  const int m = mtile * 16 + l16;    // A row for this lane
  const __bf16* erow = ebf + ((size_t)m * NS + (size_t)t) * NE;
  const __bf16* hrow = hbf + (size_t)m * NH;

  v8f acc0 = {0.f,0.f,0.f,0.f,0.f,0.f,0.f,0.f};
  v8f acc1 = acc0, acc2 = acc0, acc3 = acc0;

  for (int kb = 0; kb < NKIN; kb += 32) {
    const __bf16* arow = (kb < NE) ? (erow + kb) : (hrow + (kb - NE));
    ABFrag A;
    // lanes 0-15: K {0..7,16..23}; lanes 16-31: K {8..15,24..31}
    A.q[0] = *(const u32x4*)(arow + half * 8);
    A.q[1] = *(const u32x4*)(arow + 16 + half * 8);

    const __bf16* b0 = wg + (size_t)(nbase +  0 + l16) * NKIN + kb + half * 8;
    const __bf16* b1 = wg + (size_t)(nbase + 16 + l16) * NKIN + kb + half * 8;
    const __bf16* b2 = wg + (size_t)(nbase + 32 + l16) * NKIN + kb + half * 8;
    const __bf16* b3 = wg + (size_t)(nbase + 48 + l16) * NKIN + kb + half * 8;
    ABFrag B0, B1, B2, B3;
    B0.q[0] = *(const u32x4*)(b0); B0.q[1] = *(const u32x4*)(b0 + 16);
    B1.q[0] = *(const u32x4*)(b1); B1.q[1] = *(const u32x4*)(b1 + 16);
    B2.q[0] = *(const u32x4*)(b2); B2.q[1] = *(const u32x4*)(b2 + 16);
    B3.q[0] = *(const u32x4*)(b3); B3.q[1] = *(const u32x4*)(b3 + 16);

    acc0 = WMMA_BF16(A.v, B0.v, acc0);
    acc1 = WMMA_BF16(A.v, B1.v, acc1);
    acc2 = WMMA_BF16(A.v, B2.v, acc2);
    acc3 = WMMA_BF16(A.v, B3.v, acc3);
  }

  // C layout: VGPR r, lanes 0-15 -> M=r ; lanes 16-31 -> M=8+r ; N = l16
#define STORE_ACC(J, ACC)                                             \
  {                                                                   \
    int n = nbase + (J) * 16 + l16;                                   \
    _Pragma("unroll")                                                 \
    for (int r = 0; r < 8; ++r) {                                     \
      int mo = mtile * 16 + half * 8 + r;                             \
      gates[(size_t)mo * NG + n] = (ACC)[r];                          \
    }                                                                 \
  }
  STORE_ACC(0, acc0) STORE_ACC(1, acc1) STORE_ACC(2, acc2) STORE_ACC(3, acc3)
#undef STORE_ACC
}

// --------------------------- LSTM cell update ------------------------------
__global__ void k_lstm_cell(const float* __restrict__ gates,
                            const float* __restrict__ bg,
                            float* __restrict__ c,
                            __bf16* __restrict__ hbf,
                            __bf16* __restrict__ hs, int t) {
  int i = blockIdx.x * blockDim.x + threadIdx.x;   // B*H = 65536
  if (i >= NB * NH) return;
  int b = i >> 10, u = i & (NH - 1);
  const float* g = gates + (size_t)b * NG;
  float ig = g[u]            + bg[u];
  float fg = g[NH + u]       + bg[NH + u];
  float gg = g[2 * NH + u]   + bg[2 * NH + u];
  float og = g[3 * NH + u]   + bg[3 * NH + u];
  ig = 1.0f / (1.0f + __expf(-ig));
  fg = 1.0f / (1.0f + __expf(-fg));
  gg = tanhf(gg);
  og = 1.0f / (1.0f + __expf(-og));
  float cn = fg * c[i] + ig * gg;
  c[i] = cn;
  float h = og * tanhf(cn);
  hbf[i] = (__bf16)h;
  hs[((size_t)b * NS + t) * NH + u] = (__bf16)h;
}

// ------------------------- attention energy (WMMA) -------------------------
// energy[bs] = sum_n tanh( (hs @ Wa^T)[bs,n] ) * va[n], masked by PAD.
__global__ __launch_bounds__(256) void k_energy(
    const __bf16* __restrict__ hs,   // [B*S, H]
    const __bf16* __restrict__ wa,   // [H, H] bf16
    const float* __restrict__ va,    // [H]
    const int* __restrict__ x,       // [B*S]
    float* __restrict__ energy) {    // [B*S]
  const int lane = threadIdx.x & 31;
  const int half = lane >> 4;
  const int l16  = lane & 15;
  const int wave = blockIdx.x * (blockDim.x >> 5) + (threadIdx.x >> 5); // 0..2047
  const int rbase = wave * 16;
  const __bf16* arow = hs + (size_t)(rbase + l16) * NH;

  float esum[8];
#pragma unroll
  for (int r = 0; r < 8; ++r) esum[r] = 0.0f;

  for (int nt = 0; nt < NH / 16; ++nt) {
    v8f acc = {0.f,0.f,0.f,0.f,0.f,0.f,0.f,0.f};
    for (int kb = 0; kb < NH; kb += 32) {
      ABFrag A, Bv;
      A.q[0] = *(const u32x4*)(arow + kb + half * 8);
      A.q[1] = *(const u32x4*)(arow + kb + 16 + half * 8);
      const __bf16* brow = wa + (size_t)(nt * 16 + l16) * NH + kb + half * 8;
      Bv.q[0] = *(const u32x4*)(brow);
      Bv.q[1] = *(const u32x4*)(brow + 16);
      acc = WMMA_BF16(A.v, Bv.v, acc);
    }
    float van = va[nt * 16 + l16];
#pragma unroll
    for (int r = 0; r < 8; ++r) esum[r] += tanhf(acc[r]) * van;
  }
  // reduce over the 16 lanes of each half-wave (same M set per half)
#pragma unroll
  for (int r = 0; r < 8; ++r) {
    float v = esum[r];
    v += __shfl_xor(v, 1, 32);
    v += __shfl_xor(v, 2, 32);
    v += __shfl_xor(v, 4, 32);
    v += __shfl_xor(v, 8, 32);
    esum[r] = v;
  }
  if (l16 == 0) {
#pragma unroll
    for (int r = 0; r < 8; ++r) {
      int row = rbase + half * 8 + r;         // = b*S + s
      float e = esum[r];
      if (x[row] == 0) e = -1e10f;            // PAD mask
      energy[row] = e;
    }
  }
}

// ----------------------------- softmax over S ------------------------------
__global__ void k_softmax(const float* __restrict__ energy,
                          float* __restrict__ attn) {
  __shared__ float red[256];
  int b = blockIdx.x, tid = threadIdx.x;
  const float* e = energy + (size_t)b * NS;
  float e0 = e[tid], e1 = e[tid + 256];
  red[tid] = fmaxf(e0, e1);
  __syncthreads();
  for (int s = 128; s > 0; s >>= 1) {
    if (tid < s) red[tid] = fmaxf(red[tid], red[tid + s]);
    __syncthreads();
  }
  float m = red[0];
  __syncthreads();
  float x0 = __expf(e0 - m), x1 = __expf(e1 - m);
  red[tid] = x0 + x1;
  __syncthreads();
  for (int s = 128; s > 0; s >>= 1) {
    if (tid < s) red[tid] += red[tid + s];
    __syncthreads();
  }
  float inv = 1.0f / red[0];
  attn[(size_t)b * NS + tid]       = x0 * inv;
  attn[(size_t)b * NS + tid + 256] = x1 * inv;
}

// --------------------------- attention context -----------------------------
__global__ void k_context(const float* __restrict__ attn,
                          const __bf16* __restrict__ hs,
                          float* __restrict__ ctx) {
  int i = blockIdx.x * blockDim.x + threadIdx.x;  // B*H
  if (i >= NB * NH) return;
  int b = i >> 10, u = i & (NH - 1);
  const float* a = attn + (size_t)b * NS;
  const __bf16* h = hs + (size_t)b * NS * NH + u;
  float acc = 0.0f;
  for (int s = 0; s < NS; ++s) acc += a[s] * (float)h[(size_t)s * NH];
  ctx[i] = acc;
}

// ------------------------------ classifier ---------------------------------
__global__ void k_classify(const float* __restrict__ ctx,
                           const float* __restrict__ WV,
                           const float* __restrict__ bV,
                           float* __restrict__ out) {
  int i = blockIdx.x * blockDim.x + threadIdx.x;  // B*NCLS = 640
  if (i >= NB * 10) return;
  int b = i / 10, n = i % 10;
  const float* c = ctx + (size_t)b * NH;
  const float* w = WV + (size_t)n * NH;
  float acc = bV[n];
  for (int u = 0; u < NH; ++u) acc += c[u] * w[u];
  out[i] = acc;
}

// ------------------------------ launcher -----------------------------------
extern "C" void kernel_launch(void* const* d_in, const int* in_sizes, int n_in,
                              void* d_out, int out_size, void* d_ws, size_t ws_size,
                              hipStream_t stream) {
  const int*   x    = (const int*)d_in[0];
  const float* emb  = (const float*)d_in[1];
  const float* Wg   = (const float*)d_in[2];
  const float* bg   = (const float*)d_in[3];
  const float* Wa   = (const float*)d_in[4];
  const float* va   = (const float*)d_in[5];
  const float* WV   = (const float*)d_in[6];
  const float* bV   = (const float*)d_in[7];
  float* out = (float*)d_out;

  char* ws = (char*)d_ws;
  size_t off = 0;
  auto carve = [&](size_t bytes) {
    void* p = ws + off;
    off = (off + bytes + 255) & ~(size_t)255;
    return p;
  };
  __bf16* wg_bf = (__bf16*)carve((size_t)NG * NKIN * 2);      // 12.6 MB
  __bf16* wa_bf = (__bf16*)carve((size_t)NH * NH * 2);        //  2.1 MB
  __bf16* ebf   = (__bf16*)carve((size_t)NB * NS * NE * 2);   // 33.5 MB
  __bf16* hs_bf = (__bf16*)carve((size_t)NB * NS * NH * 2);   // 67.1 MB
  __bf16* hbf   = (__bf16*)carve((size_t)NB * NH * 2);
  float*  cst   = (float*)carve((size_t)NB * NH * 4);
  float*  gates = (float*)carve((size_t)NB * NG * 4);
  float*  energy= (float*)carve((size_t)NB * NS * 4);
  float*  attn  = (float*)carve((size_t)NB * NS * 4);
  float*  ctx   = (float*)carve((size_t)NB * NH * 4);

  // weight / embedding conversion
  {
    int n = NG * NKIN;
    k_f32_to_bf16<<<(n + 255) / 256, 256, 0, stream>>>(Wg, wg_bf, n);
  }
  {
    int n = NH * NH;
    k_f32_to_bf16<<<(n + 255) / 256, 256, 0, stream>>>(Wa, wa_bf, n);
  }
  {
    int n = NB * NS * NE;
    k_embed_gather<<<(n + 255) / 256, 256, 0, stream>>>(x, emb, ebf, n);
  }
  {
    int n = NB * NH;
    k_init_state<<<(n + 255) / 256, 256, 0, stream>>>(cst, hbf, n);
  }

  // sequential LSTM over S steps: WMMA gate GEMM + fused cell update
  for (int t = 0; t < NS; ++t) {
    k_lstm_gates<<<32, 256, 0, stream>>>(ebf, hbf, wg_bf, gates, t);
    k_lstm_cell<<<(NB * NH + 255) / 256, 256, 0, stream>>>(gates, bg, cst, hbf, hs_bf, t);
  }

  // attention energy (WMMA) + softmax + context + classifier
  k_energy<<<256, 256, 0, stream>>>(hs_bf, wa_bf, va, x, energy);
  k_softmax<<<NB, 256, 0, stream>>>(energy, attn);
  k_context<<<(NB * NH + 255) / 256, 256, 0, stream>>>(attn, hs_bf, ctx);
  k_classify<<<(NB * 10 + 255) / 256, 256, 0, stream>>>(ctx, WV, bV, out);
}
